// LSTM_6038724018759
// MI455X (gfx1250) — matrix operand order repaired and checked by reading, not compiled, not run on hardware
//
#include <hip/hip_runtime.h>
#include <hip/hip_bf16.h>
#include <math.h>
#include <stdint.h>

// Problem dims
#define kB 256
#define kT 512
#define kD 256
#define kH 1024
#define kC 512
#define kK 1280      // D + H, fused GEMM K
#define kN4 4096     // 4*H gate columns
#define NBLK 32      // persistent grid: 4 m-groups x 8 hcol-groups

typedef __attribute__((ext_vector_type(16))) __bf16 v16bf;
typedef __attribute__((ext_vector_type(8)))  __bf16 v8bf;
typedef __attribute__((ext_vector_type(8)))  float  v8f;
typedef __attribute__((ext_vector_type(4)))  unsigned tdm_u4;
typedef __attribute__((ext_vector_type(8)))  unsigned tdm_u8;

__device__ __forceinline__ float sigm(float v) {
    return 1.0f / (1.0f + __expf(-v));
}

// ---------------------------------------------------------------------------
// Pack [Wx;Wh] for 4 gates into transposed bf16 Wt[n][k], n in [0,4096), k in [0,1280)
// n = gate*1024 + hcol ; k<256 -> Wx[k][hcol], else Wh[k-256][hcol]
// ---------------------------------------------------------------------------
__global__ __launch_bounds__(256) void pack_wt_kernel(
    const float* __restrict__ wgx, const float* __restrict__ wgh,
    const float* __restrict__ wix, const float* __restrict__ wih,
    const float* __restrict__ wfx, const float* __restrict__ wfh,
    const float* __restrict__ wox, const float* __restrict__ woh,
    __bf16* __restrict__ Wt)
{
    const int n = blockIdx.y;                       // 0..4095
    const int k = blockIdx.x * 256 + threadIdx.x;   // 0..1279 (gridDim.x = 5)
    const int gate = n >> 10;
    const int hcol = n & 1023;
    const float* Wx = (gate == 0) ? wgx : (gate == 1) ? wix : (gate == 2) ? wfx : wox;
    const float* Wh = (gate == 0) ? wgh : (gate == 1) ? wih : (gate == 2) ? wfh : woh;
    float v = (k < kD) ? Wx[(size_t)k * kH + hcol]
                       : Wh[(size_t)(k - kD) * kH + hcol];
    Wt[(size_t)n * kK + k] = (__bf16)v;
}

// Pack Wph^T -> Wpht[n][k], n in [0,512), k in [0,1024)
__global__ __launch_bounds__(256) void pack_wpht_kernel(
    const float* __restrict__ wph, __bf16* __restrict__ Wpht)
{
    const int n = blockIdx.y;                       // 0..511
    const int k = blockIdx.x * 256 + threadIdx.x;   // 0..1023 (gridDim.x = 4)
    Wpht[(size_t)n * kH + k] = (__bf16)wph[(size_t)k * kC + n];
}

// Abuf0[row][k] = [x_0 | h_0 = 0] in bf16
__global__ __launch_bounds__(256) void init_abuf_kernel(
    const float* __restrict__ x, __bf16* __restrict__ Abuf0)
{
    const int row = blockIdx.y;                     // 0..255
    const int k   = blockIdx.x * 256 + threadIdx.x; // 0..1279 (gridDim.x = 5)
    float v = (k < kD) ? x[(size_t)row * kT * kD + k] : 0.0f;
    Abuf0[(size_t)row * kK + k] = (__bf16)v;
}

__global__ void zero_cnt_kernel(unsigned* cnt) {
    if (threadIdx.x == 0) *cnt = 0u;
}

// ---------------------------------------------------------------------------
// TDM descriptor issue: 2D tile [64 rows x 64 bf16] from global (row stride kK
// elements) into LDS at byte offset ldsOff, with LDS padding 128B -> 144B rows
// (pad_interval code 4 = 32 DWORDs, pad_amount code 3 = 4 DWORDs) so staged
// rows land on a 72-half stride (bank-conflict-free for WMMA fragment reads).
// Group0/Group1 per CDNA5 ISA 8.3/8.4; VADDR2/3 = NULL (<=2D tensor).
// ---------------------------------------------------------------------------
__device__ __forceinline__ void tdm_load_tile(const __bf16* gsrc, unsigned ldsOff)
{
    const unsigned long long ga = (unsigned long long)(uintptr_t)gsrc;
    tdm_u4 g0;
    g0[0] = 1u;                                        // count=1, user desc
    g0[1] = ldsOff;                                    // lds_addr
    g0[2] = (unsigned)(ga & 0xFFFFFFFFu);              // global_addr[31:0]
    g0[3] = (unsigned)((ga >> 32) & 0x01FFFFFFu)       // global_addr[56:32]
          | (2u << 30);                                // type = 2 ("image")
    tdm_u8 g1;
    g1[0] = (1u << 16)                                 // data_size = 2 bytes
          | (1u << 20)                                 // pad_enable
          | (4u << 22)                                 // pad_interval: 32 DWORDs
          | (3u << 25);                                // pad_amount: 4 DWORDs
    g1[1] = ((unsigned)kK & 0xFFFFu) << 16;            // tensor_dim0[15:0]
    g1[2] = (((unsigned)kK >> 16) & 0xFFFFu)           // tensor_dim0[31:16]
          | (((unsigned)kB & 0xFFFFu) << 16);          // tensor_dim1[15:0]
    g1[3] = (((unsigned)kB >> 16) & 0xFFFFu)           // tensor_dim1[31:16]
          | (64u << 16);                               // tile_dim0 = 64 elems
    g1[4] = 64u;                                       // tile_dim1 = 64, tile_dim2 = 0
    g1[5] = (unsigned)kK;                              // tensor_dim0_stride[31:0]
    g1[6] = 0u;                                        // stride hi / dim1_stride lo
    g1[7] = 0u;
    asm volatile("tensor_load_to_lds %0, %1" :: "s"(g0), "s"(g1) : "memory");
}

// ---------------------------------------------------------------------------
// Persistent LSTM scan. 32 WGs x 256 threads (8 wave32).
// Block (mb,nb): rows [mb*64, mb*64+64), h-cols [nb*128, nb*128+128) for all 4 gates.
// Wave w: gate = w>>1, local cols (w&1)*64 + ct*16, m-tiles 0..3, 4x4 WMMA tiles.
// A chunks DMA'd by the TDM, double-buffered to overlap with WMMA.
// ---------------------------------------------------------------------------
__global__ __launch_bounds__(256) void lstm_persistent_kernel(
    const float*  __restrict__ x,
    const __bf16* __restrict__ Wt,
    const float*  __restrict__ pbg, const float* __restrict__ pbi,
    const float*  __restrict__ pbf, const float* __restrict__ pbo,
    __bf16* Abuf0, __bf16* Abuf1,
    unsigned* barrier_cnt)
{
    const int tid   = threadIdx.x;
    const int w     = tid >> 5;
    const int lane  = tid & 31;
    const int l15   = lane & 15;
    const int lhalf = lane >> 4;
    const int mb    = blockIdx.x >> 3;   // 0..3
    const int nb    = blockIdx.x & 7;    // 0..7
    const int gate    = w >> 1;          // 0..3
    const int colbase = (w & 1) * 64;    // local col base inside the 128-col block

    __shared__ __bf16 sA[2][64][72];     // double-buffered A chunk (TDM-padded rows)
    __shared__ float  sG[64][516];       // gate pre-activations (4 gates x 128 cols)

    const unsigned ldsA0 = (unsigned)(uintptr_t)&sA[0][0][0];
    const unsigned ldsA1 = (unsigned)(uintptr_t)&sA[1][0][0];

    // Cell state kept in registers for the whole scan: 32 elements/thread.
    float creg[32];
#pragma unroll
    for (int i = 0; i < 32; ++i) creg[i] = 0.0f;

    for (int t = 0; t < kT; ++t) {
        const __bf16* Acur  = (t & 1) ? Abuf1 : Abuf0;
        __bf16*       Anext = (t & 1) ? Abuf0 : Abuf1;
        const __bf16* Arow  = Acur + (size_t)(mb * 64) * kK;

        v8f acc[4][4];
#pragma unroll
        for (int mt = 0; mt < 4; ++mt)
#pragma unroll
            for (int ct = 0; ct < 4; ++ct)
#pragma unroll
                for (int e = 0; e < 8; ++e) acc[mt][ct][e] = 0.0f;

        // Prologue: DMA chunk 0 into buffer 0.
        if (w == 0) tdm_load_tile(Arow, ldsA0);

        // K loop: 20 chunks of 64, software-pipelined against the TDM.
        for (int kc = 0; kc < kK / 64; ++kc) {
            if (w == 0) __builtin_amdgcn_s_wait_tensorcnt(0); // chunk kc landed
            __syncthreads();  // all waves: kc ready, previous buffer free
            if (w == 0 && (kc + 1) < kK / 64) {
                // DMA next chunk into the other buffer while we compute.
                tdm_load_tile(Arow + (size_t)(kc + 1) * 64,
                              ((kc + 1) & 1) ? ldsA1 : ldsA0);
            }
            const int buf = kc & 1;

#pragma unroll
            for (int k32 = 0; k32 < 2; ++k32) {
                // A fragments (ISA 7.12.2: lanes 0-15 K0-7,16-23; lanes 16-31 K8-15,24-31)
                v16bf af[4];
#pragma unroll
                for (int mt = 0; mt < 4; ++mt) {
                    const __bf16* p = &sA[buf][mt * 16 + l15][k32 * 32 + lhalf * 8];
                    v8bf lo = *(const v8bf*)p;
                    v8bf hi = *(const v8bf*)(p + 16);
                    af[mt] = __builtin_shufflevector(lo, hi,
                             0,1,2,3,4,5,6,7,8,9,10,11,12,13,14,15);
                }
#pragma unroll
                for (int ct = 0; ct < 4; ++ct) {
                    // B fragment: lane n=l15, lanes 0-15 K0-7,16-23; lanes 16-31 K8-15,24-31
                    const int n = gate * kH + nb * 128 + colbase + ct * 16 + l15;
                    const __bf16* p = Wt + (size_t)n * kK + kc * 64 + k32 * 32 + lhalf * 8;
                    v8bf lo = *(const v8bf*)p;
                    v8bf hi = *(const v8bf*)(p + 16);
                    v16bf bfr = __builtin_shufflevector(lo, hi,
                                0,1,2,3,4,5,6,7,8,9,10,11,12,13,14,15);
#pragma unroll
                    for (int mt = 0; mt < 4; ++mt) {
                        acc[mt][ct] = __builtin_amdgcn_wmma_f32_16x16x32_bf16(
                            false, af[mt], false, bfr, (short)0, acc[mt][ct],
                            false, false);
                    }
                }
            }
            __syncthreads();  // all waves done with buf before TDM may refill it
        }

        // Scatter gate tiles to LDS for cross-wave g/i/f/o combine
#pragma unroll
        for (int mt = 0; mt < 4; ++mt)
#pragma unroll
            for (int ct = 0; ct < 4; ++ct)
#pragma unroll
                for (int v = 0; v < 8; ++v) {
                    const int row = mt * 16 + v + lhalf * 8;
                    const int col = gate * 128 + colbase + ct * 16 + l15;
                    sG[row][col] = acc[mt][ct][v];
                }
        __syncthreads();

        // Activation + cell update; h_{t+1} -> Abuf[next] (bf16)
#pragma unroll 4
        for (int i = 0; i < 32; ++i) {
            const int e   = tid + i * 256;
            const int row = e >> 7;        // 0..63
            const int hl  = e & 127;       // 0..127
            const int hc  = nb * 128 + hl; // global h column
            const float gv = tanhf(sG[row][hl]        + pbg[hc]);
            const float iv = sigm (sG[row][128 + hl]  + pbi[hc]);
            const float fv = sigm (sG[row][256 + hl]  + pbf[hc]);
            const float ov = sigm (sG[row][384 + hl]  + pbo[hc]);
            const float cn = gv * iv + creg[i] * fv;
            creg[i] = cn;
            const float hv = tanhf(cn) * ov;
            Anext[(size_t)(mb * 64 + row) * kK + kD + hc] = (__bf16)hv;
        }

        // nb==0 blocks also stage x_{t+1} (f32 -> bf16) into Abuf[next]
        if (nb == 0 && (t + 1) < kT) {
            for (int j = tid; j < 64 * kD; j += 256) {
                const int row = j >> 8;
                const int k   = j & 255;
                const float v = x[(size_t)(mb * 64 + row) * kT * kD
                                  + (size_t)(t + 1) * kD + k];
                Anext[(size_t)(mb * 64 + row) * kK + k] = (__bf16)v;
            }
        }

        // Device-wide step barrier (skip after last step; kernel end syncs)
        if ((t + 1) < kT) {
            __threadfence();
            __syncthreads();
            if (tid == 0) {
                __hip_atomic_fetch_add(barrier_cnt, 1u, __ATOMIC_RELEASE,
                                       __HIP_MEMORY_SCOPE_AGENT);
                const unsigned target = (unsigned)NBLK * (unsigned)(t + 1);
                while (__hip_atomic_load(barrier_cnt, __ATOMIC_ACQUIRE,
                                         __HIP_MEMORY_SCOPE_AGENT) < target) {
                    __builtin_amdgcn_s_sleep(8);
                }
            }
            __syncthreads();
            __builtin_amdgcn_fence(__ATOMIC_ACQUIRE, "agent"); // invalidate stale WGP$ h
        }
    }
}

// ---------------------------------------------------------------------------
// out = h_T @ Wph + bp. h_T lives in Abuf0 cols [256,1280). 16 WGs x 256.
// ---------------------------------------------------------------------------
__global__ __launch_bounds__(256) void proj_kernel(
    const __bf16* __restrict__ Abuf0, const __bf16* __restrict__ Wpht,
    const float* __restrict__ pbp, float* __restrict__ out)
{
    const int tid   = threadIdx.x;
    const int w     = tid >> 5;
    const int lane  = tid & 31;
    const int l15   = lane & 15;
    const int lhalf = lane >> 4;
    const int mb    = blockIdx.x >> 2;  // 0..3 (64 rows)
    const int cb    = blockIdx.x & 3;   // 0..3 (128 out cols)

    const __bf16* Ah = Abuf0 + kD;      // h part, row stride kK

    v8f acc[4];
#pragma unroll
    for (int mt = 0; mt < 4; ++mt)
#pragma unroll
        for (int e = 0; e < 8; ++e) acc[mt][e] = 0.0f;

    const int n = cb * 128 + w * 16 + l15;  // output column for this lane
    for (int kk = 0; kk < kH; kk += 32) {
        const __bf16* pb = Wpht + (size_t)n * kH + kk + lhalf * 8;
        v8bf blo = *(const v8bf*)pb;
        v8bf bhi = *(const v8bf*)(pb + 16);
        v16bf bfr = __builtin_shufflevector(blo, bhi,
                    0,1,2,3,4,5,6,7,8,9,10,11,12,13,14,15);
#pragma unroll
        for (int mt = 0; mt < 4; ++mt) {
            const int row = mb * 64 + mt * 16 + l15;
            const __bf16* pa = Ah + (size_t)row * kK + kk + lhalf * 8;
            v8bf alo = *(const v8bf*)pa;
            v8bf ahi = *(const v8bf*)(pa + 16);
            v16bf af = __builtin_shufflevector(alo, ahi,
                       0,1,2,3,4,5,6,7,8,9,10,11,12,13,14,15);
            acc[mt] = __builtin_amdgcn_wmma_f32_16x16x32_bf16(
                false, af, false, bfr, (short)0, acc[mt], false, false);
        }
    }
#pragma unroll
    for (int mt = 0; mt < 4; ++mt)
#pragma unroll
        for (int v = 0; v < 8; ++v) {
            const int grow = mb * 64 + mt * 16 + v + lhalf * 8;
            out[(size_t)grow * kC + n] = acc[mt][v] + pbp[n];
        }
}

// ---------------------------------------------------------------------------
extern "C" void kernel_launch(void* const* d_in, const int* in_sizes, int n_in,
                              void* d_out, int out_size, void* d_ws, size_t ws_size,
                              hipStream_t stream)
{
    const float* x   = (const float*)d_in[0];
    const float* wgx = (const float*)d_in[1];
    const float* wgh = (const float*)d_in[2];
    const float* pbg = (const float*)d_in[3];
    const float* wix = (const float*)d_in[4];
    const float* wih = (const float*)d_in[5];
    const float* pbi = (const float*)d_in[6];
    const float* wfx = (const float*)d_in[7];
    const float* wfh = (const float*)d_in[8];
    const float* pbf = (const float*)d_in[9];
    const float* wox = (const float*)d_in[10];
    const float* woh = (const float*)d_in[11];
    const float* pbo = (const float*)d_in[12];
    const float* wph = (const float*)d_in[13];
    const float* pbp = (const float*)d_in[14];
    float* out = (float*)d_out;

    // Workspace layout (~12.9 MB): fused transposed weights + double-buffered A + barrier
    char* ws = (char*)d_ws;
    size_t off = 0;
    __bf16* Wt    = (__bf16*)(ws + off); off += (size_t)kN4 * kK * sizeof(__bf16); // 10.0 MB
    __bf16* Wpht  = (__bf16*)(ws + off); off += (size_t)kC  * kH * sizeof(__bf16); // 1.0 MB
    __bf16* Abuf0 = (__bf16*)(ws + off); off += (size_t)kB  * kK * sizeof(__bf16);
    __bf16* Abuf1 = (__bf16*)(ws + off); off += (size_t)kB  * kK * sizeof(__bf16);
    unsigned* cnt = (unsigned*)(ws + off); off += 256;

    zero_cnt_kernel<<<1, 64, 0, stream>>>(cnt);
    pack_wt_kernel<<<dim3(5, kN4), 256, 0, stream>>>(wgx, wgh, wix, wih,
                                                     wfx, wfh, wox, woh, Wt);
    pack_wpht_kernel<<<dim3(4, kC), 256, 0, stream>>>(wph, Wpht);
    init_abuf_kernel<<<dim3(5, kB), 256, 0, stream>>>(x, Abuf0);

    lstm_persistent_kernel<<<NBLK, 256, 0, stream>>>(x, Wt, pbg, pbi, pbf, pbo,
                                                     Abuf0, Abuf1, cnt);

    // T=512 steps: final h is written to Abuf[(511+1)&1] == Abuf0
    proj_kernel<<<16, 256, 0, stream>>>(Abuf0, Wpht, pbp, out);
}
